// MambaUpdater_67001489817848
// MI455X (gfx1250) — matrix operand order repaired and checked by reading, not compiled, not run on hardware
//
#include <hip/hip_runtime.h>
#include <hip/hip_bf16.h>

typedef __attribute__((ext_vector_type(16))) __bf16 v16bf;
typedef __attribute__((ext_vector_type(8)))  float  v8f;
typedef __attribute__((ext_vector_type(4)))  float  v4f;

#define M_TOKENS 32768   // 16 * 2048
#define LSEQ     2048
#define NBATCH   16
#define DMODEL   256
#define DINNER   512
#define DIN      1088
#define DOUT     64

// ---------------------------------------------------------------------------
// Weight convert: W[K][N] f32  ->  Wt[Npad][Kpad] bf16 (zero padded)
// ---------------------------------------------------------------------------
__global__ void wconv_kernel(const float* __restrict__ W, __bf16* __restrict__ Wt,
                             int K, int N, int Kpad, int Npad) {
    int idx = blockIdx.x * 256 + threadIdx.x;
    int total = Npad * Kpad;
    if (idx >= total) return;
    int n = idx / Kpad;
    int k = idx - n * Kpad;
    float v = (k < K && n < N) ? W[(size_t)k * N + n] : 0.0f;
    Wt[idx] = (__bf16)v;
}

// ---------------------------------------------------------------------------
// bf16 WMMA GEMM:  out[M][N] = epi( A[M][K](f32) * Wt[N][Kpad](bf16)^T )
// wave tile 16x64 (4 wmma acc), block = 4 waves -> 64x64 tile.
// If K % 32 != 0, caller must guarantee lda >= (K & ~31) + 32 (tail loads are
// unconditional; out-of-K elements are zeroed branchlessly).
// ---------------------------------------------------------------------------
enum { EPI_NONE = 0, EPI_BIAS = 1, EPI_BIAS_SOFTPLUS = 2, EPI_RESID = 3 };

template <int EPI>
__global__ __launch_bounds__(128)
void wmma_gemm(const float* __restrict__ A, const __bf16* __restrict__ Bt,
               const float* __restrict__ bias, const float* __restrict__ resid,
               float* __restrict__ out,
               int N, int K, int lda, int ldo, int Kpad) {
    const int w    = threadIdx.x >> 5;
    const int lane = threadIdx.x & 31;
    const int lm   = lane & 15;
    const int lh   = lane >> 4;            // 0: lanes 0-15, 1: lanes 16-31

    const int m0 = blockIdx.y * 64 + w * 16;
    const int n0 = blockIdx.x * 64;

    const float* Abase = A + (size_t)(m0 + lm) * lda;

    v8f acc[4] = {};
    const int Kfull = K & ~31;

    for (int k = 0; k < Kfull; k += 32) {
        // A fragment (16x32 bf16 layout): lanes 0-15 hold K {k..k+15},
        // lanes 16-31 hold K {k+16..k+31}
        const float* Ap = Abase + k + lh * 16;
        if (k + 128 < K) __builtin_prefetch(Ap + 128, 0, 3);   // WGP-scope prefetch
        v4f a0 = *(const v4f*)(Ap);
        v4f a1 = *(const v4f*)(Ap + 4);
        v4f a2 = *(const v4f*)(Ap + 8);
        v4f a3 = *(const v4f*)(Ap + 12);
        v16bf af;
#pragma unroll
        for (int i = 0; i < 4; i++) {
            af[i]      = (__bf16)a0[i];
            af[4 + i]  = (__bf16)a1[i];
            af[8 + i]  = (__bf16)a2[i];
            af[12 + i] = (__bf16)a3[i];
        }
#pragma unroll
        for (int ns = 0; ns < 4; ns++) {
            const __bf16* Bp = Bt + (size_t)(n0 + ns * 16 + lm) * Kpad + k + lh * 16;
            v16bf bf = *(const v16bf*)Bp;
            acc[ns] = __builtin_amdgcn_wmma_f32_16x16x32_bf16(
                false, af, false, bf, (short)0, acc[ns], false, false);
        }
    }

    if (K & 31) {           // tail (e.g. dt proj K=16): unconditional loads,
        const int k = Kfull;                 // branchless zeroing (v_cndmask)
        const float* Ap = Abase + k + lh * 16;
        v4f a0 = *(const v4f*)(Ap);
        v4f a1 = *(const v4f*)(Ap + 4);
        v4f a2 = *(const v4f*)(Ap + 8);
        v4f a3 = *(const v4f*)(Ap + 12);
        v16bf af;
#pragma unroll
        for (int i = 0; i < 4; i++) {
            int k0 = k + lh * 16 + i;        // ke for chunk0
            int k1 = k + lh * 16 + 8 + i;    // ke for chunk1
            af[i]      = (__bf16)((k0 < K)     ? a0[i] : 0.0f);
            af[4 + i]  = (__bf16)((k0 + 4 < K) ? a1[i] : 0.0f);
            af[8 + i]  = (__bf16)((k1 < K)     ? a2[i] : 0.0f);
            af[12 + i] = (__bf16)((k1 + 4 < K) ? a3[i] : 0.0f);
        }
#pragma unroll
        for (int ns = 0; ns < 4; ns++) {
            const __bf16* Bp = Bt + (size_t)(n0 + ns * 16 + lm) * Kpad + k + lh * 16;
            v16bf bf = *(const v16bf*)Bp;   // pad region of Wt is zero-filled
            acc[ns] = __builtin_amdgcn_wmma_f32_16x16x32_bf16(
                false, af, false, bf, (short)0, acc[ns], false, false);
        }
    }

    // Epilogue. C layout: column = lm, rows = m0 + lh*8 + r, r = 0..7
#pragma unroll
    for (int ns = 0; ns < 4; ns++) {
        int nn = n0 + ns * 16 + lm;
        if (nn >= N) continue;
        float bv = (EPI == EPI_BIAS || EPI == EPI_BIAS_SOFTPLUS) ? bias[nn] : 0.0f;
#pragma unroll
        for (int r = 0; r < 8; r++) {
            int m = m0 + lh * 8 + r;
            size_t o = (size_t)m * ldo + nn;
            float v = acc[ns][r] + bv;
            if (EPI == EPI_BIAS_SOFTPLUS) v = (v > 15.0f) ? v : log1pf(__expf(v));
            if (EPI == EPI_RESID)         v += resid[o];
            out[o] = v;
        }
    }
}

// ---------------------------------------------------------------------------
// LayerNorm: one wave (32 lanes) per row of width W
// ---------------------------------------------------------------------------
template <int W>
__global__ __launch_bounds__(256)
void ln_kernel(const float* __restrict__ in, const float* __restrict__ g,
               const float* __restrict__ b, float* __restrict__ out, int rows) {
    int wid  = (blockIdx.x * blockDim.x + threadIdx.x) >> 5;
    int lane = threadIdx.x & 31;
    if (wid >= rows) return;
    const float* rp = in + (size_t)wid * W;
    constexpr int PER = W / 32;
    float v[PER];
    float s = 0.0f;
#pragma unroll
    for (int i = 0; i < PER; i++) { v[i] = rp[lane + 32 * i]; s += v[i]; }
#pragma unroll
    for (int m = 16; m >= 1; m >>= 1) s += __shfl_xor(s, m, 32);
    float mean = s * (1.0f / W);
    float q = 0.0f;
#pragma unroll
    for (int i = 0; i < PER; i++) { float d = v[i] - mean; q += d * d; }
#pragma unroll
    for (int m = 16; m >= 1; m >>= 1) q += __shfl_xor(q, m, 32);
    float rstd = rsqrtf(q * (1.0f / W) + 1e-5f);
    float* op = out + (size_t)wid * W;
#pragma unroll
    for (int i = 0; i < PER; i++) {
        int c = lane + 32 * i;
        op[c] = (v[i] - mean) * rstd * g[c] + b[c];
    }
}

// ---------------------------------------------------------------------------
// Causal depthwise conv (K=4) + SiLU, reads xc half of xz
// ---------------------------------------------------------------------------
__global__ __launch_bounds__(256)
void conv_silu_kernel(const float* __restrict__ xz, const float* __restrict__ cw,
                      const float* __restrict__ cb, float* __restrict__ xc) {
    int idx = blockIdx.x * 256 + threadIdx.x;      // over M_TOKENS * 512
    int d   = idx & 511;
    int row = idx >> 9;
    int l   = row & (LSEQ - 1);
    const float* w = cw + d * 4;
    float acc = cb[d];
#pragma unroll
    for (int k = 0; k < 4; k++) {
        int ls = l - 3 + k;
        if (ls >= 0) acc += w[k] * xz[(size_t)(row - 3 + k) * 1024 + d];
    }
    xc[(size_t)row * 512 + d] = acc / (1.0f + __expf(-acc));
}

// ---------------------------------------------------------------------------
// Selective scan: 1 thread per (batch, channel). 16-state recurrence in VGPRs.
// B/C rows (shared by all 512 channels of a batch) double-buffered in LDS.
// Fuses skip (u*D) and gating (* silu(z)).
// ---------------------------------------------------------------------------
__global__ __launch_bounds__(256)
void scan_kernel(const float* __restrict__ dtb, const float* __restrict__ xc,
                 const float* __restrict__ xz, const float* __restrict__ xdbl,
                 const float* __restrict__ Alog, const float* __restrict__ Dp,
                 float* __restrict__ ymul) {
    __shared__ float Bs[2][32][16];
    __shared__ float Cs[2][32][16];

    const int b    = blockIdx.x >> 1;
    const int half = blockIdx.x & 1;
    const int d    = half * 256 + threadIdx.x;
    const size_t tok0 = (size_t)b * LSEQ;

    float h[16], An[16];
#pragma unroll
    for (int n = 0; n < 16; n++) { h[n] = 0.0f; An[n] = -__expf(Alog[d * 16 + n]); }
    const float Dd = Dp[d];

    auto load_win = [&](int w, int buf) {
        int l0 = w * 32;
        for (int t = threadIdx.x; t < 512; t += 256) {
            int j = t >> 4, n = t & 15;
            size_t base = (tok0 + l0 + j) * 48;
            Bs[buf][j][n] = xdbl[base + 16 + n];
            Cs[buf][j][n] = xdbl[base + 32 + n];
        }
    };

    load_win(0, 0);
    __syncthreads();

    for (int w = 0; w < LSEQ / 32; w++) {
        int cur = w & 1;
        if (w + 1 < LSEQ / 32) load_win(w + 1, cur ^ 1);
        int l0 = w * 32;
        for (int j = 0; j < 32; j++) {
            size_t tok = tok0 + l0 + j;
            float dt = dtb[tok * 512 + d];
            float u  = xc[tok * 512 + d];
            float z  = xz[tok * 1024 + 512 + d];
            float dtu = dt * u;
            float y0 = 0, y1 = 0, y2 = 0, y3 = 0;
#pragma unroll
            for (int n = 0; n < 16; n += 4) {
                float e0 = __expf(dt * An[n]);     h[n]     = e0 * h[n]     + Bs[cur][j][n]     * dtu; y0 += Cs[cur][j][n]     * h[n];
                float e1 = __expf(dt * An[n + 1]); h[n + 1] = e1 * h[n + 1] + Bs[cur][j][n + 1] * dtu; y1 += Cs[cur][j][n + 1] * h[n + 1];
                float e2 = __expf(dt * An[n + 2]); h[n + 2] = e2 * h[n + 2] + Bs[cur][j][n + 2] * dtu; y2 += Cs[cur][j][n + 2] * h[n + 2];
                float e3 = __expf(dt * An[n + 3]); h[n + 3] = e3 * h[n + 3] + Bs[cur][j][n + 3] * dtu; y3 += Cs[cur][j][n + 3] * h[n + 3];
            }
            float y  = (y0 + y1) + (y2 + y3);
            float sz = z / (1.0f + __expf(-z));
            ymul[tok * 512 + d] = (y + u * Dd) * sz;
        }
        __syncthreads();
    }
}

// ---------------------------------------------------------------------------
// Host orchestration
// ---------------------------------------------------------------------------
static inline dim3 gemm_grid(int Npad) { return dim3(Npad / 64, M_TOKENS / 64, 1); }

extern "C" void kernel_launch(void* const* d_in, const int* in_sizes, int n_in,
                              void* d_out, int out_size, void* d_ws, size_t ws_size,
                              hipStream_t stream) {
    (void)in_sizes; (void)n_in; (void)out_size; (void)ws_size;

    const float* z_t    = (const float*)d_in[0];
    const float* W_in   = (const float*)d_in[1];
    const float* b_in   = (const float*)d_in[2];
    const float* ln_g   = (const float*)d_in[3];
    const float* ln_b   = (const float*)d_in[4];
    const float* W_inp  = (const float*)d_in[5];
    const float* conv_w = (const float*)d_in[6];
    const float* conv_b = (const float*)d_in[7];
    const float* W_xp   = (const float*)d_in[8];
    const float* W_dt   = (const float*)d_in[9];
    const float* b_dt   = (const float*)d_in[10];
    const float* A_log  = (const float*)d_in[11];
    const float* D_skip = (const float*)d_in[12];
    const float* W_out  = (const float*)d_in[13];
    const float* W_outp = (const float*)d_in[14];
    const float* b_outp = (const float*)d_in[15];
    const float* lno_g  = (const float*)d_in[16];
    const float* lno_b  = (const float*)d_in[17];
    float* out = (float*)d_out;

    // ----- workspace layout (deterministic) -----
    char* ws = (char*)d_ws;
    size_t cur = 0;
    auto take = [&](size_t bytes) -> size_t {
        size_t o = cur; cur += (bytes + 255) & ~(size_t)255; return o;
    };
    __bf16* wt_in     = (__bf16*)(ws + take((size_t)256 * 1088 * 2));
    __bf16* wt_inp[2] = { (__bf16*)(ws + take((size_t)1024 * 256 * 2)),
                          (__bf16*)(ws + take((size_t)1024 * 256 * 2)) };
    __bf16* wt_xp[2]  = { (__bf16*)(ws + take((size_t)64 * 512 * 2)),
                          (__bf16*)(ws + take((size_t)64 * 512 * 2)) };
    __bf16* wt_dt[2]  = { (__bf16*)(ws + take((size_t)512 * 32 * 2)),
                          (__bf16*)(ws + take((size_t)512 * 32 * 2)) };
    __bf16* wt_out[2] = { (__bf16*)(ws + take((size_t)256 * 512 * 2)),
                          (__bf16*)(ws + take((size_t)256 * 512 * 2)) };
    __bf16* wt_outp   = (__bf16*)(ws + take((size_t)64 * 256 * 2));

    float* x    = (float*)(ws + take((size_t)M_TOKENS * 256 * 4));
    float* xn   = (float*)(ws + take((size_t)M_TOKENS * 256 * 4));
    float* xz   = (float*)(ws + take((size_t)M_TOKENS * 1024 * 4));
    float* xc   = (float*)(ws + take((size_t)M_TOKENS * 512 * 4));
    float* xdbl = (float*)(ws + take((size_t)M_TOKENS * 48 * 4));
    float* dtb  = (float*)(ws + take((size_t)M_TOKENS * 512 * 4));
    float* ymul = (float*)(ws + take((size_t)M_TOKENS * 512 * 4));
    float* uo   = (float*)(ws + take((size_t)M_TOKENS * 64 * 4));

    // ----- weight convert/transpose to bf16 [N][Kpad] -----
    auto wconv = [&](const float* W, __bf16* Wt, int K, int N, int Kpad, int Npad) {
        int total = Npad * Kpad;
        wconv_kernel<<<(total + 255) / 256, 256, 0, stream>>>(W, Wt, K, N, Kpad, Npad);
    };
    wconv(W_in, wt_in, 1088, 256, 1088, 256);
    for (int i = 0; i < 2; i++) {
        wconv(W_inp + (size_t)i * 256 * 1024, wt_inp[i], 256, 1024, 256, 1024);
        wconv(W_xp  + (size_t)i * 512 * 48,   wt_xp[i],  512, 48,   512, 64);
        wconv(W_dt  + (size_t)i * 16 * 512,   wt_dt[i],  16,  512,  32,  512);
        wconv(W_out + (size_t)i * 512 * 256,  wt_out[i], 512, 256,  512, 256);
    }
    wconv(W_outp, wt_outp, 256, 64, 256, 64);

    const int ln_blocks = (M_TOKENS * 32) / 256;

    // ----- input projection: x = z_t @ W_in + b_in -----
    wmma_gemm<EPI_BIAS><<<gemm_grid(256), 128, 0, stream>>>(
        z_t, wt_in, b_in, nullptr, x, 256, 1088, 1088, 256, 1088);

    // ----- Mamba layers -----
    for (int i = 0; i < 2; i++) {
        ln_kernel<256><<<ln_blocks, 256, 0, stream>>>(
            x, ln_g + i * 256, ln_b + i * 256, xn, M_TOKENS);

        wmma_gemm<EPI_NONE><<<gemm_grid(1024), 128, 0, stream>>>(
            xn, wt_inp[i], nullptr, nullptr, xz, 1024, 256, 256, 1024, 256);

        conv_silu_kernel<<<(M_TOKENS * 512) / 256, 256, 0, stream>>>(
            xz, conv_w + (size_t)i * 512 * 4, conv_b + (size_t)i * 512, xc);

        wmma_gemm<EPI_NONE><<<gemm_grid(64), 128, 0, stream>>>(
            xc, wt_xp[i], nullptr, nullptr, xdbl, 48, 512, 512, 48, 512);

        wmma_gemm<EPI_BIAS_SOFTPLUS><<<gemm_grid(512), 128, 0, stream>>>(
            xdbl, wt_dt[i], b_dt + (size_t)i * 512, nullptr, dtb, 512, 16, 48, 512, 32);

        scan_kernel<<<NBATCH * 2, 256, 0, stream>>>(
            dtb, xc, xz, xdbl, A_log + (size_t)i * 512 * 16,
            D_skip + (size_t)i * 512, ymul);

        wmma_gemm<EPI_RESID><<<gemm_grid(256), 128, 0, stream>>>(
            ymul, wt_out[i], nullptr, x, x, 256, 512, 512, 256, 512);
    }

    // ----- output projection + final layernorm -----
    wmma_gemm<EPI_BIAS><<<gemm_grid(64), 128, 0, stream>>>(
        x, wt_outp, b_outp, nullptr, uo, 64, 256, 256, 64, 256);

    ln_kernel<64><<<ln_blocks, 256, 0, stream>>>(uo, lno_g, lno_b, out, M_TOKENS);
}
